// trajectory2seq_40948218200388
// MI455X (gfx1250) — matrix operand order repaired and checked by reading, not compiled, not run on hardware
//
#include <hip/hip_runtime.h>
#include <hip/hip_bf16.h>

typedef __bf16 bf16_t;
typedef __attribute__((ext_vector_type(16))) __bf16 v16bf;
typedef __attribute__((ext_vector_type(8)))  float  v8f;

union V16U { v16bf v; uint4 q[2]; __bf16 h[16]; };
union V8FU { v8f v; float f[8]; };

__device__ __forceinline__ v8f wmma_bf16f32(v16bf a, v16bf b, v8f c) {
  return __builtin_amdgcn_wmma_f32_16x16x32_bf16(false, a, false, b, (short)0, c, false, false);
}

// gfx1250 async global->LDS copy (ASYNCcnt-tracked), per cdna5_isa §15.18 / 08_async_tensor.
// LDS byte address = low 32 bits of the flat pointer.
__device__ __forceinline__ void async_ld_lds_b32(void* lds_ptr, const void* gptr) {
  unsigned lds_off = (unsigned)(unsigned long long)lds_ptr;
  asm volatile("global_load_async_to_lds_b32 %0, %1, off"
               :: "v"(lds_off), "v"(gptr) : "memory");
}
__device__ __forceinline__ void wait_asynccnt0() {
  asm volatile("s_wait_asynccnt 0x0" ::: "memory");
}

constexpr int Hdim = 256;
constexpr int Tlen = 512;
constexpr int Bsz  = 256;
constexpr int Llen = 64;
constexpr int DICT = 29;
constexpr int ENC_BLOCKS = 16;

__device__ __forceinline__ float sigmoidf_(float x) { return 1.f / (1.f + __expf(-x)); }

// ---------------------------------------------------------------- init
__global__ void init_kernel(float* out, int out_n, bf16_t* hA, int* tok, int* bar) {
  int idx = blockIdx.x * blockDim.x + threadIdx.x;
  int stride = gridDim.x * blockDim.x;
  for (int i = idx; i < out_n; i += stride) {
    float v = 0.f;
    if (i < Bsz * Llen * DICT && (i % (Llen * DICT)) == 0) v = 1.f;
    out[i] = v;
  }
  for (int i = idx; i < Bsz * Hdim; i += stride) hA[i] = (bf16_t)0.f;
  for (int i = idx; i < Bsz; i += stride) tok[i] = 1;  // <sos>
  if (idx < 2) bar[idx] = 0;
}

// ---------------------------------------------------------------- f32 -> bf16
__global__ void cvt_kernel(const float* __restrict__ src, bf16_t* __restrict__ dst, int n) {
  int idx = blockIdx.x * blockDim.x + threadIdx.x;
  int stride = gridDim.x * blockDim.x;
  for (int i = idx; i < n; i += stride) dst[i] = (bf16_t)src[i];
}

// ---------------------------------------------------------------- encoder (persistent, grid barrier per step)
__global__ __launch_bounds__(256)
void enc_kernel(const float* __restrict__ x,       // (B,2,T)
                const float* __restrict__ wih,     // (768,2) f32
                const float* __restrict__ bih,
                const float* __restrict__ bhh,
                const bf16_t* __restrict__ whh_bf, // (768,256) bf16
                bf16_t* __restrict__ hA,           // ping (B,H) bf16  (pre-zeroed)
                bf16_t* __restrict__ hB,           // pong
                float*  __restrict__ hout_f32,     // final hidden (B,H) f32
                bf16_t* __restrict__ vout,         // (B,T,H) bf16
                int* __restrict__ bar) {
  __shared__ bf16_t ldsW[48 * 256];  // 3 gates x 16 cols of Whh rows
  __shared__ float xs[2 * Bsz];      // this step's x: [0:256)=ch0, [256:512)=ch1
  const int tid = threadIdx.x;
  const int cb = blockIdx.x;       // column tile 0..15
  const int wave = tid >> 5;
  const int lane = tid & 31;
  const int lo = lane & 15, hi = lane >> 4;

  for (int i = tid; i < 48 * 256; i += 256) {
    int rr = i >> 8, kk = i & 255;
    int g = rr >> 4, cc = rr & 15;
    ldsW[i] = whh_bf[(g * 256 + cb * 16 + cc) * 256 + kk];
  }
  __syncthreads();

  const int c = cb * 16 + lo;  // this lane's output column
  const float wr0 = wih[(0 * 256 + c) * 2 + 0], wr1 = wih[(0 * 256 + c) * 2 + 1];
  const float wz0 = wih[(1 * 256 + c) * 2 + 0], wz1 = wih[(1 * 256 + c) * 2 + 1];
  const float wn0 = wih[(2 * 256 + c) * 2 + 0], wn1 = wih[(2 * 256 + c) * 2 + 1];
  const float br  = bih[c] + bhh[c];
  const float bz  = bih[256 + c] + bhh[256 + c];
  const float bin = bih[512 + c];
  const float bhn = bhh[512 + c];

  float hreg[2][8];
  #pragma unroll
  for (int ts = 0; ts < 2; ++ts)
    #pragma unroll
    for (int e = 0; e < 8; ++e) hreg[ts][e] = 0.f;

  int* cnt = bar;
  int* gen = bar + 1;

  #pragma unroll 1
  for (int t = 0; t < Tlen; ++t) {
    const bf16_t* hin = (t & 1) ? hB : hA;
    bf16_t* hout = (t & 1) ? hA : hB;

    // stage this timestep's x column into LDS via async global->LDS copies
    async_ld_lds_b32(&xs[tid],        x + tid * (2 * Tlen) + t);
    async_ld_lds_b32(&xs[256 + tid],  x + tid * (2 * Tlen) + Tlen + t);
    wait_asynccnt0();
    __syncthreads();

    #pragma unroll 1
    for (int ts = 0; ts < 2; ++ts) {
      const int rbase = (wave * 2 + ts) * 16;
      V8FU a0, a1, a2;
      #pragma unroll
      for (int e = 0; e < 8; ++e) { a0.f[e] = 0.f; a1.f[e] = 0.f; a2.f[e] = 0.f; }
      const bf16_t* arow = hin + (rbase + lo) * Hdim;
      // software-pipelined A-tile loads (one K-block ahead)
      V16U ac, an;
      ac.q[0] = *(const uint4*)(arow + hi * 16);
      ac.q[1] = *(const uint4*)(arow + hi * 16 + 8);
      #pragma unroll
      for (int kb = 0; kb < 8; ++kb) {
        if (kb < 7) {
          const int k1 = (kb + 1) * 32;
          an.q[0] = *(const uint4*)(arow + k1 + hi * 16);
          an.q[1] = *(const uint4*)(arow + k1 + hi * 16 + 8);
        }
        const int k0 = kb * 32;
        V16U b0, b1, b2;
        const bf16_t* w0p = &ldsW[(0 * 16 + lo) * 256 + k0 + hi * 16];
        const bf16_t* w1p = &ldsW[(1 * 16 + lo) * 256 + k0 + hi * 16];
        const bf16_t* w2p = &ldsW[(2 * 16 + lo) * 256 + k0 + hi * 16];
        b0.q[0] = *(const uint4*)(w0p); b0.q[1] = *(const uint4*)(w0p + 8);
        b1.q[0] = *(const uint4*)(w1p); b1.q[1] = *(const uint4*)(w1p + 8);
        b2.q[0] = *(const uint4*)(w2p); b2.q[1] = *(const uint4*)(w2p + 8);
        a0.v = wmma_bf16f32(ac.v, b0.v, a0.v);
        a1.v = wmma_bf16f32(ac.v, b1.v, a1.v);
        a2.v = wmma_bf16f32(ac.v, b2.v, a2.v);
        ac = an;
      }
      #pragma unroll
      for (int e = 0; e < 8; ++e) {
        const int row = rbase + e + hi * 8;
        const float x0 = xs[row];
        const float x1 = xs[256 + row];
        const float r = sigmoidf_(a0.f[e] + x0 * wr0 + x1 * wr1 + br);
        const float z = sigmoidf_(a1.f[e] + x0 * wz0 + x1 * wz1 + bz);
        const float n = tanhf(x0 * wn0 + x1 * wn1 + bin + r * (a2.f[e] + bhn));
        const float hn = (1.f - z) * n + z * hreg[ts][e];
        hreg[ts][e] = hn;
        hout[row * Hdim + c] = (bf16_t)hn;
        vout[((size_t)row * Tlen + t) * Hdim + c] = (bf16_t)hn;
      }
    }
    // grid-wide barrier (16 co-resident blocks)
    __syncthreads();
    if (tid == 0) {
      __threadfence();
      volatile int* vg = (volatile int*)gen;
      int g0 = *vg;
      if (atomicAdd(cnt, 1) == ENC_BLOCKS - 1) {
        *(volatile int*)cnt = 0;
        __threadfence();
        atomicAdd(gen, 1);
      } else {
        while (*vg == g0) __builtin_amdgcn_s_sleep(2);
      }
      __threadfence();
    }
    __syncthreads();
  }
  #pragma unroll
  for (int ts = 0; ts < 2; ++ts) {
    const int rbase = (wave * 2 + ts) * 16;
    #pragma unroll
    for (int e = 0; e < 8; ++e) hout_f32[(rbase + e + hi * 8) * Hdim + c] = hreg[ts][e];
  }
}

// ---------------------------------------------------------------- 1/max(||v||, eps)
__global__ __launch_bounds__(256)
void vnorm_kernel(const bf16_t* __restrict__ v, float* __restrict__ vni) {
  int wid = blockIdx.x * 8 + (threadIdx.x >> 5);  // row 0..B*T-1
  int lane = threadIdx.x & 31;
  const bf16_t* row = v + (size_t)wid * Hdim;
  union { uint4 u; __bf16 h[8]; } X;
  X.u = *(const uint4*)(row + lane * 8);
  float s = 0.f;
  #pragma unroll
  for (int e = 0; e < 8; ++e) { float f = (float)X.h[e]; s += f * f; }
  #pragma unroll
  for (int off = 16; off; off >>= 1) s += __shfl_xor(s, off, 32);
  if (lane == 0) vni[wid] = 1.f / fmaxf(sqrtf(s), 1e-8f);
}

// ---------------------------------------------------------------- decoder GRU step (WMMA, 2 GEMMs fused)
__global__ __launch_bounds__(256)
void dec_gru(const bf16_t* __restrict__ emb_bf,  // (29,256)
             const int* __restrict__ tok,
             const float* __restrict__ dbih, const float* __restrict__ dbhh,
             const bf16_t* __restrict__ wih_bf,  // (768,256)
             const bf16_t* __restrict__ whh_bf,  // (768,256)
             const bf16_t* __restrict__ hin_bf,
             const float* __restrict__ hin_f32,
             bf16_t* __restrict__ hout_bf,
             float*  __restrict__ hout_f32,
             bf16_t* __restrict__ comb) {        // (B,512): writes [:,0:256] = q
  __shared__ bf16_t ldsWi[48 * 256];
  __shared__ bf16_t ldsWh[48 * 256];
  const int tid = threadIdx.x;
  const int cb = blockIdx.x;
  const int wave = tid >> 5;
  const int lane = tid & 31;
  const int lo = lane & 15, hi = lane >> 4;

  for (int i = tid; i < 48 * 256; i += 256) {
    int rr = i >> 8, kk = i & 255;
    int g = rr >> 4, cc = rr & 15;
    int src = (g * 256 + cb * 16 + cc) * 256 + kk;
    ldsWi[i] = wih_bf[src];
    ldsWh[i] = whh_bf[src];
  }
  __syncthreads();

  const int c = cb * 16 + lo;
  const float br  = dbih[c] + dbhh[c];
  const float bz  = dbih[256 + c] + dbhh[256 + c];
  const float bin = dbih[512 + c];
  const float bhn = dbhh[512 + c];

  #pragma unroll 1
  for (int ts = 0; ts < 2; ++ts) {
    const int rbase = (wave * 2 + ts) * 16;
    const int tk = tok[rbase + lo];
    const bf16_t* erow = emb_bf + tk * Hdim;
    const bf16_t* hrow = hin_bf + (rbase + lo) * Hdim;
    V8FU ir, iz, inn, hr, hz, hn;
    #pragma unroll
    for (int e = 0; e < 8; ++e) {
      ir.f[e] = 0.f; iz.f[e] = 0.f; inn.f[e] = 0.f;
      hr.f[e] = 0.f; hz.f[e] = 0.f; hn.f[e] = 0.f;
    }
    // software-pipelined A loads for both GEMMs
    V16U aec, ahc, aen, ahn2;
    aec.q[0] = *(const uint4*)(erow + hi * 16);
    aec.q[1] = *(const uint4*)(erow + hi * 16 + 8);
    ahc.q[0] = *(const uint4*)(hrow + hi * 16);
    ahc.q[1] = *(const uint4*)(hrow + hi * 16 + 8);
    #pragma unroll
    for (int kb = 0; kb < 8; ++kb) {
      if (kb < 7) {
        const int k1 = (kb + 1) * 32;
        aen.q[0] = *(const uint4*)(erow + k1 + hi * 16);
        aen.q[1] = *(const uint4*)(erow + k1 + hi * 16 + 8);
        ahn2.q[0] = *(const uint4*)(hrow + k1 + hi * 16);
        ahn2.q[1] = *(const uint4*)(hrow + k1 + hi * 16 + 8);
      }
      const int k0 = kb * 32;
      #pragma unroll
      for (int g = 0; g < 3; ++g) {
        V16U bi, bh;
        const bf16_t* wip = &ldsWi[(g * 16 + lo) * 256 + k0 + hi * 16];
        const bf16_t* whp = &ldsWh[(g * 16 + lo) * 256 + k0 + hi * 16];
        bi.q[0] = *(const uint4*)(wip); bi.q[1] = *(const uint4*)(wip + 8);
        bh.q[0] = *(const uint4*)(whp); bh.q[1] = *(const uint4*)(whp + 8);
        if (g == 0) { ir.v  = wmma_bf16f32(aec.v, bi.v, ir.v);  hr.v = wmma_bf16f32(ahc.v, bh.v, hr.v); }
        else if (g == 1) { iz.v = wmma_bf16f32(aec.v, bi.v, iz.v); hz.v = wmma_bf16f32(ahc.v, bh.v, hz.v); }
        else { inn.v = wmma_bf16f32(aec.v, bi.v, inn.v); hn.v = wmma_bf16f32(ahc.v, bh.v, hn.v); }
      }
      aec = aen; ahc = ahn2;
    }
    #pragma unroll
    for (int e = 0; e < 8; ++e) {
      const int row = rbase + e + hi * 8;
      const float hold = hin_f32[row * Hdim + c];
      const float r = sigmoidf_(ir.f[e] + hr.f[e] + br);
      const float z = sigmoidf_(iz.f[e] + hz.f[e] + bz);
      const float n = tanhf(inn.f[e] + bin + r * (hn.f[e] + bhn));
      const float hv = (1.f - z) * n + z * hold;
      hout_f32[row * Hdim + c] = hv;
      hout_bf[row * Hdim + c] = (bf16_t)hv;
      comb[row * 512 + c] = (bf16_t)hv;
    }
  }
}

// ---------------------------------------------------------------- attention: cos, softmax, context
__global__ __launch_bounds__(256)
void dec_attn(const float* __restrict__ hq,    // new hidden (B,H) f32
              const bf16_t* __restrict__ v,    // (B,T,H)
              const float* __restrict__ vni,   // (B,T)
              float* __restrict__ attn_out,    // (B,T,L+2)
              bf16_t* __restrict__ comb,       // (B,512): writes [:,256:512] = a
              int l) {
  __shared__ float q[256];
  __shared__ float wgt[512];
  __shared__ float red[8];
  __shared__ float sc[2];
  const int b = blockIdx.x, tid = threadIdx.x;
  const int wave = tid >> 5, lane = tid & 31;

  float qv = hq[b * Hdim + tid];
  q[tid] = qv;
  float p = qv * qv;
  #pragma unroll
  for (int off = 16; off; off >>= 1) p += __shfl_xor(p, off, 32);
  if (lane == 0) red[wave] = p;
  __syncthreads();
  if (tid == 0) {
    float s = 0.f;
    for (int i = 0; i < 8; ++i) s += red[i];
    sc[0] = 1.f / fmaxf(sqrtf(s), 1e-8f);
  }
  __syncthreads();
  const float qninv = sc[0];

  const bf16_t* vb = v + (size_t)b * Tlen * Hdim;
  for (int i = 0; i < 64; ++i) {
    const int t = wave * 64 + i;
    union { uint4 u; __bf16 h[8]; } X;
    X.u = *(const uint4*)(vb + (size_t)t * Hdim + lane * 8);
    float d = 0.f;
    #pragma unroll
    for (int e = 0; e < 8; ++e) d += (float)X.h[e] * q[lane * 8 + e];
    #pragma unroll
    for (int off = 16; off; off >>= 1) d += __shfl_xor(d, off, 32);
    if (lane == 0) wgt[t] = d * vni[b * Tlen + t] * qninv;
  }
  __syncthreads();

  // softmax over 512 (each thread owns t = tid, tid+256)
  float m0 = fmaxf(wgt[tid], wgt[tid + 256]);
  #pragma unroll
  for (int off = 16; off; off >>= 1) m0 = fmaxf(m0, __shfl_xor(m0, off, 32));
  if (lane == 0) red[wave] = m0;
  __syncthreads();
  if (tid == 0) {
    float m = red[0];
    for (int i = 1; i < 8; ++i) m = fmaxf(m, red[i]);
    sc[1] = m;
  }
  __syncthreads();
  const float mx = sc[1];
  float e0 = __expf(wgt[tid] - mx), e1 = __expf(wgt[tid + 256] - mx);
  float p2 = e0 + e1;
  #pragma unroll
  for (int off = 16; off; off >>= 1) p2 += __shfl_xor(p2, off, 32);
  if (lane == 0) red[wave] = p2;
  __syncthreads();
  if (tid == 0) {
    float s = 0.f;
    for (int i = 0; i < 8; ++i) s += red[i];
    sc[0] = 1.f / s;
  }
  __syncthreads();
  const float inv = sc[0];
  e0 *= inv; e1 *= inv;
  wgt[tid] = e0; wgt[tid + 256] = e1;
  attn_out[((size_t)b * Tlen + tid) * (Llen + 2) + l] = e0;
  attn_out[((size_t)b * Tlen + tid + 256) * (Llen + 2) + l] = e1;
  __syncthreads();

  // context a[c] = sum_t wgt[t] * v[b,t,c]   (coalesced over c)
  float a = 0.f;
  for (int t = 0; t < Tlen; ++t) a += wgt[t] * (float)vb[(size_t)t * Hdim + tid];
  comb[b * 512 + 256 + tid] = (bf16_t)a;
}

// ---------------------------------------------------------------- hid = comb @ attn_W^T + b (K=512 WMMA)
__global__ __launch_bounds__(256)
void dec_hid(const bf16_t* __restrict__ comb,     // (256,512)
             const bf16_t* __restrict__ attnW_bf, // (256,512)
             const float* __restrict__ attn_b,
             bf16_t* __restrict__ hid) {          // (256,256)
  const int wave = threadIdx.x >> 5;
  const int lane = threadIdx.x & 31;
  const int lo = lane & 15, hi = lane >> 4;
  const int wid = blockIdx.x * 8 + wave;    // 0..255
  const int rbase = (wid >> 4) * 16, cbase = (wid & 15) * 16;

  V8FU acc;
  #pragma unroll
  for (int e = 0; e < 8; ++e) acc.f[e] = 0.f;
  const bf16_t* arow = comb + (rbase + lo) * 512;
  const bf16_t* brow = attnW_bf + (cbase + lo) * 512;
  V16U ac, bc, an, bn;
  ac.q[0] = *(const uint4*)(arow + hi * 16);
  ac.q[1] = *(const uint4*)(arow + hi * 16 + 8);
  bc.q[0] = *(const uint4*)(brow + hi * 16);
  bc.q[1] = *(const uint4*)(brow + hi * 16 + 8);
  #pragma unroll
  for (int kb = 0; kb < 16; ++kb) {
    if (kb < 15) {
      const int k1 = (kb + 1) * 32;
      an.q[0] = *(const uint4*)(arow + k1 + hi * 16);
      an.q[1] = *(const uint4*)(arow + k1 + hi * 16 + 8);
      bn.q[0] = *(const uint4*)(brow + k1 + hi * 16);
      bn.q[1] = *(const uint4*)(brow + k1 + hi * 16 + 8);
    }
    acc.v = wmma_bf16f32(ac.v, bc.v, acc.v);
    ac = an; bc = bn;
  }
  const float bias = attn_b[cbase + lo];
  #pragma unroll
  for (int e = 0; e < 8; ++e)
    hid[(rbase + e + hi * 8) * Hdim + cbase + lo] = (bf16_t)(acc.f[e] + bias);
}

// ---------------------------------------------------------------- logits + argmax + vec_out row
__global__ __launch_bounds__(32)
void dec_out(const bf16_t* __restrict__ hid, const float* __restrict__ fcW,
             const float* __restrict__ fcb, float* __restrict__ vec_out,
             int* __restrict__ tok, int l) {
  const int b = blockIdx.x, lane = threadIdx.x;
  float logit = -3.4e38f;
  if (lane < DICT) {
    float s = fcb[lane];
    const bf16_t* hr = hid + b * Hdim;
    const float* wr = fcW + lane * Hdim;
    for (int k = 0; k < Hdim; ++k) s += (float)hr[k] * wr[k];
    logit = s;
    vec_out[((size_t)b * Llen + l) * DICT + lane] = s;
  }
  float best = logit; int bi = lane;
  #pragma unroll
  for (int off = 16; off; off >>= 1) {
    float ob = __shfl_xor(best, off, 32);
    int oi = __shfl_xor(bi, off, 32);
    if (ob > best || (ob == best && oi < bi)) { best = ob; bi = oi; }
  }
  if (lane == 0) tok[b] = bi;
}

// ---------------------------------------------------------------- copy final hidden
__global__ void copy_h(const float* __restrict__ src, float* __restrict__ dst) {
  int i = blockIdx.x * blockDim.x + threadIdx.x;
  dst[i] = src[i];
}

// ---------------------------------------------------------------- host
extern "C" void kernel_launch(void* const* d_in, const int* in_sizes, int n_in,
                              void* d_out, int out_size, void* d_ws, size_t ws_size,
                              hipStream_t stream) {
  const float* x    = (const float*)d_in[0];
  const float* emb  = (const float*)d_in[1];
  const float* eWih = (const float*)d_in[2];
  const float* eWhh = (const float*)d_in[3];
  const float* eBih = (const float*)d_in[4];
  const float* eBhh = (const float*)d_in[5];
  const float* dWih = (const float*)d_in[6];
  const float* dWhh = (const float*)d_in[7];
  const float* dBih = (const float*)d_in[8];
  const float* dBhh = (const float*)d_in[9];
  const float* aW   = (const float*)d_in[10];
  const float* aB   = (const float*)d_in[11];
  const float* fcW  = (const float*)d_in[12];
  const float* fcB  = (const float*)d_in[13];

  char* ws = (char*)d_ws;
  bf16_t* emb_bf  = (bf16_t*)(ws + 0);
  bf16_t* eWhh_bf = (bf16_t*)(ws + 16384);
  bf16_t* dWih_bf = (bf16_t*)(ws + 409600);
  bf16_t* dWhh_bf = (bf16_t*)(ws + 802816);
  bf16_t* aW_bf   = (bf16_t*)(ws + 1196032);
  bf16_t* hbf0    = (bf16_t*)(ws + 1458176);
  bf16_t* hbf1    = (bf16_t*)(ws + 1589248);
  float*  hf0     = (float*) (ws + 1720320);
  float*  hf1     = (float*) (ws + 1982464);
  bf16_t* combp   = (bf16_t*)(ws + 2244608);
  bf16_t* hidp    = (bf16_t*)(ws + 2506752);
  int*    tokp    = (int*)   (ws + 2637824);
  int*    barp    = (int*)   (ws + 2638848);
  float*  vnip    = (float*) (ws + 2639104);
  bf16_t* vbf     = (bf16_t*)(ws + 3163392);

  float* out = (float*)d_out;
  float* vec_out = out;                                // B*L*DICT = 475136
  float* hfin    = out + Bsz * Llen * DICT;            // 65536
  float* attn    = hfin + Bsz * Hdim;                  // B*T*(L+2)

  init_kernel<<<512, 256, 0, stream>>>(out, out_size, hbf0, tokp, barp);

  cvt_kernel<<<64, 256, 0, stream>>>(emb,  emb_bf,  DICT * Hdim);
  cvt_kernel<<<256, 256, 0, stream>>>(eWhh, eWhh_bf, 3 * Hdim * Hdim);
  cvt_kernel<<<256, 256, 0, stream>>>(dWih, dWih_bf, 3 * Hdim * Hdim);
  cvt_kernel<<<256, 256, 0, stream>>>(dWhh, dWhh_bf, 3 * Hdim * Hdim);
  cvt_kernel<<<256, 256, 0, stream>>>(aW,   aW_bf,   Hdim * 2 * Hdim);

  enc_kernel<<<ENC_BLOCKS, 256, 0, stream>>>(x, eWih, eBih, eBhh, eWhh_bf,
                                             hbf0, hbf1, hf0, vbf, barp);

  vnorm_kernel<<<(Bsz * Tlen) / 8, 256, 0, stream>>>(vbf, vnip);

  for (int l = 0; l < Llen - 1; ++l) {
    bf16_t* hin_bf  = (l & 1) ? hbf1 : hbf0;
    bf16_t* hout_bf = (l & 1) ? hbf0 : hbf1;
    float*  hin_f   = (l & 1) ? hf1 : hf0;
    float*  hout_f  = (l & 1) ? hf0 : hf1;
    dec_gru<<<16, 256, 0, stream>>>(emb_bf, tokp, dBih, dBhh, dWih_bf, dWhh_bf,
                                    hin_bf, hin_f, hout_bf, hout_f, combp);
    dec_attn<<<256, 256, 0, stream>>>(hout_f, vbf, vnip, attn, combp, l);
    dec_hid<<<32, 256, 0, stream>>>(combp, aW_bf, aB, hidp);
    dec_out<<<256, 32, 0, stream>>>(hidp, fcW, fcB, vec_out, tokp, l + 1);
  }
  // after l = 62 (even), final hidden is in hf1
  copy_h<<<256, 256, 0, stream>>>(hf1, hfin);
}
